// GRU_84121229460100
// MI455X (gfx1250) — compile-verified, matrix operand order checked by reading, not run on hardware
//
#include <hip/hip_runtime.h>
#include <hip/hip_bf16.h>

#define B_   256
#define T_   128
#define D_   1024
#define H_   1024
#define G3H  3072

typedef __bf16 bf16;
typedef __attribute__((ext_vector_type(16))) __bf16 v16bf;
typedef __attribute__((ext_vector_type(8)))  __bf16 v8bf;
typedef __attribute__((ext_vector_type(8)))  float  v8f;

// Build a 16-element bf16 fragment from two contiguous 16-byte chunks.
// Layout per CDNA5 ISA (16-bit A/B matrices, wave32):
//   lanes 0-15  hold K = [k0, k0+8) and [k0+16, k0+24)
//   lanes 16-31 hold K = [k0+8, k0+16) and [k0+24, k0+32)
// byteoff already includes 2*khalf with khalf = (lane>=16 ? 8 : 0).
// Uniform base + unsigned 32-bit byte offset -> saddr global_load_b128 form.
static __device__ __forceinline__ v16bf load_frag(const bf16* __restrict__ base,
                                                  unsigned byteoff) {
  const char* p = reinterpret_cast<const char*>(base);
  v8bf lo = *reinterpret_cast<const v8bf*>(p + byteoff);
  v8bf hi = *reinterpret_cast<const v8bf*>(p + byteoff + 32u);
  return __builtin_shufflevector(lo, hi, 0, 1, 2, 3, 4, 5, 6, 7,
                                 8, 9, 10, 11, 12, 13, 14, 15);
}

#define WMMA_BF16(A, Bm, C) \
  __builtin_amdgcn_wmma_f32_16x16x32_bf16(false, (A), false, (Bm), (short)0, (C), false, false)

__global__ void cvt_f32_bf16(const float* __restrict__ in, bf16* __restrict__ out, long n) {
  long i = (long)blockIdx.x * blockDim.x + threadIdx.x;
  long stride = (long)gridDim.x * blockDim.x;
  for (; i < n; i += stride) out[i] = (bf16)in[i];
}

__global__ void zero_h(float* __restrict__ hf, bf16* __restrict__ hb) {
  int i = blockIdx.x * blockDim.x + threadIdx.x;   // launched with exactly B_*H_ threads
  hf[i] = 0.0f;
  hb[i] = (bf16)0.0f;
}

// xg[b*T+t, 3H] = x_bf16 @ w_ih_bf16^T + b_ih, f32 output.
// One wave -> 16(M) x 64(N) tile, software-pipelined (double-buffered frags).
// Block = 8 waves = 16 x 512; grid = (M/16, N/512).
__global__ void __launch_bounds__(256)
xg_gemm(const bf16* __restrict__ xb, const bf16* __restrict__ wb,
        const float* __restrict__ b_ih, float* __restrict__ xg) {
  const int lane  = threadIdx.x & 31;
  const int wave  = threadIdx.x >> 5;
  const int mrow  = lane & 15;
  const int khalf = (lane >> 4) << 3;
  const unsigned row0 = blockIdx.x * 16u;
  const unsigned col0 = blockIdx.y * 512u + wave * 64u;

  // Per-lane unsigned BYTE offsets off uniform bases (saddr-friendly).
  const unsigned offA  = ((row0 + mrow) * (unsigned)D_ + khalf) * 2u;
  const unsigned offB0 = ((col0 +  0u + mrow) * (unsigned)D_ + khalf) * 2u;
  const unsigned offB1 = ((col0 + 16u + mrow) * (unsigned)D_ + khalf) * 2u;
  const unsigned offB2 = ((col0 + 32u + mrow) * (unsigned)D_ + khalf) * 2u;
  const unsigned offB3 = ((col0 + 48u + mrow) * (unsigned)D_ + khalf) * 2u;

  v8f acc0 = {}, acc1 = {}, acc2 = {}, acc3 = {};

  // Prologue: fragments for chunk 0.
  v16bf a  = load_frag(xb, offA);
  v16bf b0 = load_frag(wb, offB0);
  v16bf b1 = load_frag(wb, offB1);
  v16bf b2 = load_frag(wb, offB2);
  v16bf b3 = load_frag(wb, offB3);

  // 32 k-chunks of 32 elements (64 bytes each); pipeline next-load vs. WMMA.
#pragma unroll 4
  for (int kk = 1; kk < D_ / 32; ++kk) {
    const unsigned d = (unsigned)kk * 64u;
    v16bf an  = load_frag(xb, offA + d);
    v16bf bn0 = load_frag(wb, offB0 + d);
    v16bf bn1 = load_frag(wb, offB1 + d);
    v16bf bn2 = load_frag(wb, offB2 + d);
    v16bf bn3 = load_frag(wb, offB3 + d);
    acc0 = WMMA_BF16(a, b0, acc0);
    acc1 = WMMA_BF16(a, b1, acc1);
    acc2 = WMMA_BF16(a, b2, acc2);
    acc3 = WMMA_BF16(a, b3, acc3);
    a = an; b0 = bn0; b1 = bn1; b2 = bn2; b3 = bn3;
  }
  // Drain.
  acc0 = WMMA_BF16(a, b0, acc0);
  acc1 = WMMA_BF16(a, b1, acc1);
  acc2 = WMMA_BF16(a, b2, acc2);
  acc3 = WMMA_BF16(a, b3, acc3);

  // C/D layout: VGPR i -> M = i + 8*(lane>=16), N = lane%16.
  const int mbase = (lane >> 4) << 3;
  v8f accs[4] = {acc0, acc1, acc2, acc3};
#pragma unroll
  for (int j = 0; j < 4; ++j) {
    const int col = col0 + 16 * j + (lane & 15);
    const float bias = b_ih[col];
#pragma unroll
    for (int i = 0; i < 8; ++i) {
      const long r = (long)row0 + mbase + i;
      xg[r * (long)G3H + col] = accs[j][i] + bias;
    }
  }
}

// One GRU timestep. One wave computes a 16(B) x 16(H) tile of h_new,
// accumulating the r/z/n gate tiles (rows j, H+j, 2H+j of w_hh) with a shared
// A fragment; double-buffered like xg_gemm.
// grid = 128 blocks * 8 waves = 1024 tiles = (B/16)*(H/16).
__global__ void __launch_bounds__(256)
gru_step(const bf16* __restrict__ hprev_b, const float* __restrict__ hprev_f,
         const bf16* __restrict__ whh, const float* __restrict__ b_hh,
         const float* __restrict__ xg, const int t,
         float* __restrict__ hout_f, bf16* __restrict__ hout_b,
         float* __restrict__ out, float* __restrict__ hx) {
  const int lane  = threadIdx.x & 31;
  const int wave  = threadIdx.x >> 5;
  const int tile  = blockIdx.x * 8 + wave;  // 0..1023
  const unsigned tn = tile & 63;            // H tile index
  const unsigned tm = tile >> 6;            // B tile index
  const int mrow  = lane & 15;
  const int khalf = (lane >> 4) << 3;

  const unsigned offA  = ((tm * 16u + mrow) * (unsigned)H_ + khalf) * 2u;
  const unsigned offBr = ((              tn * 16u + mrow) * (unsigned)H_ + khalf) * 2u;
  const unsigned offBz = (((unsigned)H_     + tn * 16u + mrow) * (unsigned)H_ + khalf) * 2u;
  const unsigned offBn = (((unsigned)2 * H_ + tn * 16u + mrow) * (unsigned)H_ + khalf) * 2u;

  v8f cr = {}, cz = {}, cn = {};

  v16bf a  = load_frag(hprev_b, offA);
  v16bf br = load_frag(whh, offBr);
  v16bf bz = load_frag(whh, offBz);
  v16bf bn = load_frag(whh, offBn);

#pragma unroll 4
  for (int kk = 1; kk < H_ / 32; ++kk) {
    const unsigned d = (unsigned)kk * 64u;
    v16bf an  = load_frag(hprev_b, offA + d);
    v16bf brn = load_frag(whh, offBr + d);
    v16bf bzn = load_frag(whh, offBz + d);
    v16bf bnn = load_frag(whh, offBn + d);
    cr = WMMA_BF16(a, br, cr);
    cz = WMMA_BF16(a, bz, cz);
    cn = WMMA_BF16(a, bn, cn);
    a = an; br = brn; bz = bzn; bn = bnn;
  }
  cr = WMMA_BF16(a, br, cr);
  cz = WMMA_BF16(a, bz, cz);
  cn = WMMA_BF16(a, bn, cn);

  const int mbase = (lane >> 4) << 3;
  const int col = tn * 16 + (lane & 15);
  const float bhr = b_hh[col];
  const float bhz = b_hh[H_ + col];
  const float bhn = b_hh[2 * H_ + col];
#pragma unroll
  for (int i = 0; i < 8; ++i) {
    const int b = tm * 16 + mbase + i;
    const long xrow = ((long)b * T_ + t) * (long)G3H;
    const float xr = xg[xrow + col];
    const float xz = xg[xrow + H_ + col];
    const float xn = xg[xrow + 2 * H_ + col];
    const float r = 1.0f / (1.0f + __expf(-(xr + cr[i] + bhr)));
    const float z = 1.0f / (1.0f + __expf(-(xz + cz[i] + bhz)));
    const float nn = tanhf(xn + r * (cn[i] + bhn));
    const float hp = hprev_f[(long)b * H_ + col];
    const float hnew = (1.0f - z) * nn + z * hp;
    hout_f[(long)b * H_ + col] = hnew;
    hout_b[(long)b * H_ + col] = (bf16)hnew;
    out[((long)b * T_ + t) * (long)H_ + col] = hnew;
    if (hx) hx[(long)b * H_ + col] = hnew;
  }
}

extern "C" void kernel_launch(void* const* d_in, const int* in_sizes, int n_in,
                              void* d_out, int out_size, void* d_ws, size_t ws_size,
                              hipStream_t stream) {
  (void)in_sizes; (void)n_in; (void)out_size; (void)ws_size;
  const float* x    = (const float*)d_in[0];  // [B,T,D]
  const float* w_ih = (const float*)d_in[1];  // [3H,D]
  const float* w_hh = (const float*)d_in[2];  // [3H,H]
  const float* b_ih = (const float*)d_in[3];  // [3H]
  const float* b_hh = (const float*)d_in[4];  // [3H]

  char* ws = (char*)d_ws;
  size_t off = 0;
  bf16* xb   = (bf16*)(ws + off); off += (size_t)B_ * T_ * D_ * sizeof(bf16);
  bf16* wihb = (bf16*)(ws + off); off += (size_t)G3H * D_ * sizeof(bf16);
  bf16* whhb = (bf16*)(ws + off); off += (size_t)G3H * H_ * sizeof(bf16);
  float* xg  = (float*)(ws + off); off += (size_t)B_ * T_ * G3H * sizeof(float);
  float* hf0 = (float*)(ws + off); off += (size_t)B_ * H_ * sizeof(float);
  float* hf1 = (float*)(ws + off); off += (size_t)B_ * H_ * sizeof(float);
  bf16* hb0  = (bf16*)(ws + off);  off += (size_t)B_ * H_ * sizeof(bf16);
  bf16* hb1  = (bf16*)(ws + off);  off += (size_t)B_ * H_ * sizeof(bf16);
  float* hf[2] = {hf0, hf1};
  bf16*  hb[2] = {hb0, hb1};

  // Precision conversion (one pass, bandwidth-trivial vs. the GEMMs).
  cvt_f32_bf16<<<4096, 256, 0, stream>>>(x, xb, (long)B_ * T_ * D_);
  cvt_f32_bf16<<<1024, 256, 0, stream>>>(w_ih, wihb, (long)G3H * D_);
  cvt_f32_bf16<<<1024, 256, 0, stream>>>(w_hh, whhb, (long)G3H * H_);
  zero_h<<<(B_ * H_) / 256, 256, 0, stream>>>(hf[0], hb[0]);

  // Phase 1: batched input projection, 32768 x 3072 x 1024 bf16 WMMA GEMM.
  xg_gemm<<<dim3((B_ * T_) / 16, G3H / 512), 256, 0, stream>>>(xb, wihb, b_ih, xg);

  // Phase 2: sequential scan; w_hh (6 MB bf16) stays resident in L2.
  float* out = (float*)d_out;                     // [B,T,H]
  float* hx  = out + (size_t)B_ * T_ * H_;        // [B,H]
  for (int t = 0; t < T_; ++t) {
    const int cur = t & 1, nxt = cur ^ 1;
    gru_step<<<128, 256, 0, stream>>>(hb[cur], hf[cur], whhb, b_hh, xg, t,
                                      hf[nxt], hb[nxt], out,
                                      (t == T_ - 1) ? hx : nullptr);
  }
}
